// Efficient_Spiking_Transformer_44530220925342
// MI455X (gfx1250) — 16-bit and 8-bit matrix operand order repaired and checked by reading, compiled, not run on hardware
//
#include <hip/hip_runtime.h>


typedef __attribute__((ext_vector_type(16))) _Float16 v16h;
typedef __attribute__((ext_vector_type(8)))  _Float16 v8h;
typedef __attribute__((ext_vector_type(8)))  float    v8f;
typedef __attribute__((ext_vector_type(8)))  int      v8i;
typedef __attribute__((ext_vector_type(4)))  int      v4i;

#define SHUF16(lo, hi) __builtin_shufflevector((lo), (hi), 0,1,2,3,4,5,6,7,8,9,10,11,12,13,14,15)
#define SHUF8(lo, hi)  __builtin_shufflevector((lo), (hi), 0,1,2,3,4,5,6,7)

__device__ __forceinline__ int spike_i(float x) {
  x = fminf(fmaxf(x, 0.0f), 8.0f);
  return (int)floorf(x + 0.5f);
}
__device__ __forceinline__ float spikef(float x) { return (float)spike_i(x) * 0.125f; }

__device__ __forceinline__ unsigned char fp8_of_spike(int si) {
  if (si == 0) return 0;
  int p = 31 - __clz(si);
  return (unsigned char)(((4 + p) << 3) | ((si << (3 - p)) & 7));
}

__device__ __forceinline__ int rowbase(int t, int br, int win, int n0) {
  int b = win >> 1, sub = win & 1;
  int h, wc;
  if (br == 0) { h = n0 >> 4;              wc = sub * 16; }
  else         { h = sub * 16 + (n0 >> 5); wc = n0 & 31;  }
  return (t * 4 + b) * 1024 + h * 32 + wc;
}

__global__ void cvt_kernel(const float* __restrict__ s, _Float16* __restrict__ d, int n) {
  int i = blockIdx.x * 256 + threadIdx.x;
  if (i < n) d[i] = (_Float16)s[i];
}

__global__ void bn_prep_kernel(const float* __restrict__ p, float* __restrict__ sc,
                               float* __restrict__ sh, int c) {
  int i = blockIdx.x * 256 + threadIdx.x;
  if (i < c) {
    float g = p[i], b = p[c + i], m = p[2 * c + i], v = p[3 * c + i];
    float inv = g * rsqrtf(v + 1e-5f);
    sc[i] = inv;
    sh[i] = b - m * inv;
  }
}

__global__ void spike_transpose_kernel(const float* __restrict__ x, _Float16* __restrict__ xs) {
  __shared__ float tile[32][33];
  int tb = blockIdx.z, c0 = blockIdx.y * 32, hw0 = blockIdx.x * 32;
  int tx = threadIdx.x & 31, ty = threadIdx.x >> 5;
#pragma unroll
  for (int r = 0; r < 4; ++r) {
    int cc = ty + r * 8;
    tile[cc][tx] = x[((size_t)(tb * 512 + c0 + cc) << 10) + hw0 + tx];
  }
  __syncthreads();
#pragma unroll
  for (int r = 0; r < 4; ++r) {
    int hh = ty + r * 8;
    xs[((size_t)(tb * 1024 + hw0 + hh) << 9) + c0 + tx] = (_Float16)spikef(tile[tx][hh]);
  }
}

template <int MODE>
__global__ void __launch_bounds__(128) wgemm_kernel(
    const _Float16* __restrict__ A, const _Float16* __restrict__ Bt,
    const float* __restrict__ bsc, const float* __restrict__ bsh,
    void* __restrict__ out, int Kdim, int Ndim, int Mdim) {
  __shared__ __align__(16) unsigned char stg[4][16 * 24];
  int wave = threadIdx.x >> 5, lane = threadIdx.x & 31;
  int lrow = lane & 15, lhi = lane >> 4;
  int m0 = blockIdx.x * 64;
  int n0 = blockIdx.y * 256 + wave * 64;

  v8f acc[4][4] = {};
#pragma unroll 1
  for (int k0 = 0; k0 < Kdim; k0 += 32) {
    v16h af[4], bf[4];
#pragma unroll
    for (int i = 0; i < 4; ++i) {
      const _Float16* p = A + (size_t)(m0 + i * 16 + lrow) * Kdim + k0 + lhi * 16;
      v8h lo = *(const v8h*)p, hi = *(const v8h*)(p + 8);
      af[i] = SHUF16(lo, hi);
    }
#pragma unroll
    for (int j = 0; j < 4; ++j)
      bf[j] = *(const v16h*)(Bt + (size_t)(n0 + j * 16 + lrow) * Kdim + k0 + lhi * 16);
#pragma unroll
    for (int i = 0; i < 4; ++i)
#pragma unroll
      for (int j = 0; j < 4; ++j)
        acc[i][j] = __builtin_amdgcn_wmma_f32_16x16x32_f16(false, af[i], false, bf[j],
                                                           (short)0, acc[i][j], false, false);
  }

#pragma unroll 1
  for (int i = 0; i < 4; ++i) {
#pragma unroll 1
    for (int j = 0; j < 4; ++j) {
      int ch = n0 + j * 16 + lrow;
      float sc = bsc[ch], sh = bsh[ch];
      if (MODE == 0) {
        unsigned char* sb = stg[wave];
        __builtin_amdgcn_wave_barrier();
#pragma unroll
        for (int r = 0; r < 8; ++r)
          sb[(r + 8 * lhi) * 24 + lrow] = fp8_of_spike(spike_i(acc[i][j][r] * sc + sh));
        __builtin_amdgcn_wave_barrier();
        int row = lane >> 1, hb = lane & 1;
        unsigned long long d = *(const unsigned long long*)(sb + row * 24 + hb * 8);
        *(unsigned long long*)((unsigned char*)out +
            (size_t)(m0 + i * 16 + row) * Ndim + n0 + j * 16 + hb * 8) = d;
      } else if (MODE == 1) {
        v8h y;
#pragma unroll
        for (int r = 0; r < 8; ++r) y[r] = (_Float16)spikef(acc[i][j][r] * sc + sh);
        *(v8h*)((_Float16*)out + (size_t)ch * Mdim + m0 + i * 16 + lhi * 8) = y;
      } else {
        int mrow = m0 + i * 16 + lhi * 8;
        int tb = mrow >> 10, hw = mrow & 1023;
        v8f y;
#pragma unroll
        for (int r = 0; r < 8; ++r) y[r] = acc[i][j][r] * sc + sh;
        *(v8f*)((float*)out + ((size_t)(tb * 512 + ch) << 10) + hw) = y;
      }
    }
  }
}

__global__ void __launch_bounds__(128) attn_kernel(
    const unsigned char* __restrict__ q8, const unsigned char* __restrict__ k8,
    const _Float16* __restrict__ vT, _Float16* __restrict__ xa) {
  __shared__ __align__(32) _Float16 sS[4][16 * 40];
  int wave = threadIdx.x >> 5, lane = threadIdx.x & 31;
  int lrow = lane & 15, lhi = lane >> 4;
  int gw = blockIdx.x * 4 + wave;
  int rowblk = gw & 31;
  int head = (gw >> 5) & 3;
  int win  = (gw >> 7) & 7;
  int br   = (gw >> 10) & 1;
  int t    = (gw >> 11) & 3;
  int cq = br * 256 + head * 64;
  int cv = br * 1024 + head * 256;
  int mq = rowbase(t, br, win, rowblk * 16);
  _Float16* sb = sS[wave];

  v8i aq;
  {
    const unsigned char* p = q8 + (size_t)(mq + lrow) * 512 + cq + lhi * 16;
    const int* p0 = (const int*)p;
    const int* p1 = (const int*)(p + 8);
    const int* p2 = (const int*)(p + 32);
    const int* p3 = (const int*)(p + 40);
    aq[0] = p0[0]; aq[1] = p0[1]; aq[2] = p1[0]; aq[3] = p1[1];
    aq[4] = p2[0]; aq[5] = p2[1]; aq[6] = p3[0]; aq[7] = p3[1];
  }

  v8f accO[16] = {};
#pragma unroll 1
  for (int kc = 0; kc < 16; ++kc) {
    int mka = rowbase(t, br, win, kc * 32);
    int mkb = rowbase(t, br, win, kc * 32 + 16);
    const unsigned char* pa = k8 + (size_t)(mka + lrow) * 512 + cq + lhi * 16;
    const unsigned char* pb = k8 + (size_t)(mkb + lrow) * 512 + cq + lhi * 16;
    v8i bk0 = SHUF8(*(const v4i*)pa, *(const v4i*)(pa + 32));
    v8i bk1 = SHUF8(*(const v4i*)pb, *(const v4i*)(pb + 32));
    v8f s0 = {}, s1 = {};
    s0 = __builtin_amdgcn_wmma_f32_16x16x64_fp8_fp8(aq, bk0, (short)0, s0, false, false);
    s1 = __builtin_amdgcn_wmma_f32_16x16x64_fp8_fp8(aq, bk1, (short)0, s1, false, false);

    __builtin_amdgcn_wave_barrier();
#pragma unroll
    for (int r = 0; r < 8; ++r) {
      int row = r + 8 * lhi;
      sb[row * 40 + lrow]      = (_Float16)s0[r];
      sb[row * 40 + 16 + lrow] = (_Float16)s1[r];
    }
    __builtin_amdgcn_wave_barrier();
    v16h aS;
    {
      const _Float16* p = sb + lrow * 40 + lhi * 16;
      aS = SHUF16(*(const v8h*)p, *(const v8h*)(p + 8));
    }
    __builtin_amdgcn_wave_barrier();

#pragma unroll
    for (int nt = 0; nt < 16; ++nt) {
      const _Float16* vp = vT + ((size_t)(cv + nt * 16 + lrow) << 14) + (lhi ? mkb : mka);
      v16h bv = *(const v16h*)vp;
      accO[nt] = __builtin_amdgcn_wmma_f32_16x16x32_f16(false, aS, false, bv,
                                                        (short)0, accO[nt], false, false);
    }
  }

#pragma unroll 1
  for (int nt = 0; nt < 16; ++nt) {
    __builtin_amdgcn_wave_barrier();
#pragma unroll
    for (int r = 0; r < 8; ++r)
      sb[(r + 8 * lhi) * 40 + lrow] = (_Float16)spikef(accO[nt][r] * 0.25f);
    __builtin_amdgcn_wave_barrier();
    int row = lane >> 1, hb = lane & 1;
    v8h d = *(const v8h*)(sb + row * 40 + hb * 8);
    *(v8h*)(xa + ((size_t)(mq + row) << 11) + cv + nt * 16 + hb * 8) = d;
  }
}

extern "C" void kernel_launch(void* const* d_in, const int* in_sizes, int n_in,
                              void* d_out, int out_size, void* d_ws, size_t ws_size,
                              hipStream_t stream) {
  (void)in_sizes; (void)n_in; (void)out_size; (void)ws_size;
  const float* x   = (const float*)d_in[0];
  const float* qw  = (const float*)d_in[1];
  const float* qbn = (const float*)d_in[2];
  const float* kw  = (const float*)d_in[3];
  const float* kbn = (const float*)d_in[4];
  const float* vw  = (const float*)d_in[5];
  const float* vbn = (const float*)d_in[6];
  const float* pw  = (const float*)d_in[7];
  const float* pbn = (const float*)d_in[8];

  char* ws = (char*)d_ws;
  _Float16* xs  = (_Float16*)(ws + 0);
  _Float16* wqf = (_Float16*)(ws + 16777216);
  _Float16* wkf = (_Float16*)(ws + 17301504);
  _Float16* wvf = (_Float16*)(ws + 17825792);
  _Float16* wpf = (_Float16*)(ws + 19922944);
  float* bnp = (float*)(ws + 22020096);
  float *qs = bnp,        *qh = bnp + 512;
  float *ks = bnp + 1024, *kh = bnp + 1536;
  float *vs = bnp + 2048, *vh = bnp + 4096;
  float *ps = bnp + 6144, *ph = bnp + 6656;
  unsigned char* q8 = (unsigned char*)(ws + 22048768);
  unsigned char* k8 = (unsigned char*)(ws + 30437376);
  _Float16* vT = (_Float16*)(ws + 38825984);
  _Float16* xa = (_Float16*)(ws + 105934848);

  cvt_kernel<<<1024, 256, 0, stream>>>(qw, wqf, 262144);
  cvt_kernel<<<1024, 256, 0, stream>>>(kw, wkf, 262144);
  cvt_kernel<<<4096, 256, 0, stream>>>(vw, wvf, 1048576);
  cvt_kernel<<<4096, 256, 0, stream>>>(pw, wpf, 1048576);
  bn_prep_kernel<<<2, 256, 0, stream>>>(qbn, qs, qh, 512);
  bn_prep_kernel<<<2, 256, 0, stream>>>(kbn, ks, kh, 512);
  bn_prep_kernel<<<8, 256, 0, stream>>>(vbn, vs, vh, 2048);
  bn_prep_kernel<<<2, 256, 0, stream>>>(pbn, ps, ph, 512);

  spike_transpose_kernel<<<dim3(32, 16, 16), 256, 0, stream>>>(x, xs);

  wgemm_kernel<0><<<dim3(256, 2), 128, 0, stream>>>(xs, wqf, qs, qh, q8, 512, 512, 16384);
  wgemm_kernel<0><<<dim3(256, 2), 128, 0, stream>>>(xs, wkf, ks, kh, k8, 512, 512, 16384);
  wgemm_kernel<1><<<dim3(256, 8), 128, 0, stream>>>(xs, wvf, vs, vh, vT, 512, 2048, 16384);

  attn_kernel<<<2048, 128, 0, stream>>>(q8, k8, vT, xa);

  wgemm_kernel<2><<<dim3(256, 2), 128, 0, stream>>>(xa, wpf, ps, ph, d_out, 2048, 512, 16384);
}
